// SparseMoEBlock_67765993997290
// MI455X (gfx1250) — matrix operand order repaired and checked by reading, not compiled, not run on hardware
//
#include <hip/hip_runtime.h>
#include <hip/hip_bf16.h>
#include <math.h>

// Problem constants (match reference).
#define Bc    8
#define Sc    8192
#define Dc    512
#define Ec    16
#define DFFc  512
#define Kc    1024        // S/E * CAPACITY tokens per expert
#define MT    64          // token rows per workgroup in the fused MLP

typedef __attribute__((ext_vector_type(16))) __bf16 v16bf;
typedef __attribute__((ext_vector_type(8)))  __bf16 v8bf;
typedef __attribute__((ext_vector_type(8)))  float  v8f;

union V16 { v16bf v; v8bf h[2]; };

static __device__ __forceinline__ unsigned int pack_bf16x2(float a, float b) {
  __bf16 ha = (__bf16)a, hb = (__bf16)b;
  unsigned short ua = __builtin_bit_cast(unsigned short, ha);
  unsigned short ub = __builtin_bit_cast(unsigned short, hb);
  return (unsigned int)ua | ((unsigned int)ub << 16);
}

// tanh-approx GELU, branchless: 0.5*v*(1+tanh(u)) == v * sigmoid(2u).
static __device__ __forceinline__ float gelu_tanh(float v) {
  const float c0 = 0.7978845608028654f;          // sqrt(2/pi)
  float u = c0 * (v + 0.044715f * v * v * v);
  return v / (1.0f + __expf(-2.0f * u));
}

// ---------------------------------------------------------------------------
// Kernel 1: fp32 -> bf16 weight conversion (w1 and w2), E*D*DFF each.
// ---------------------------------------------------------------------------
__global__ void moe_cvt_weights(const float* __restrict__ w1,
                                const float* __restrict__ w2,
                                unsigned short* __restrict__ w1b,
                                unsigned short* __restrict__ w2b) {
  const int n = Ec * Dc * DFFc;
  int i = blockIdx.x * blockDim.x + threadIdx.x;
  if (i < n) {
    __bf16 a = (__bf16)w1[i];
    w1b[i] = __builtin_bit_cast(unsigned short, a);
    __bf16 b = (__bf16)w2[i];
    w2b[i] = __builtin_bit_cast(unsigned short, b);
  }
}

// ---------------------------------------------------------------------------
// Kernel 2: router.  affinity[b,e,s] = softmax_e(x[b,s,:] @ gw[:,e]).
// Block = 16 tokens x 16 experts.
// ---------------------------------------------------------------------------
__global__ __launch_bounds__(256) void moe_router(const float* __restrict__ x,
                                                  const float* __restrict__ gw,
                                                  float* __restrict__ aff) {
  const int e  = threadIdx.x & 15;
  const int tl = threadIdx.x >> 4;
  const int sblk = blockIdx.x & (Sc / 16 - 1);
  const int b    = blockIdx.x / (Sc / 16);
  const int s = sblk * 16 + tl;

  const float* xrow = x + ((size_t)b * Sc + s) * Dc;
  float acc = 0.0f;
  #pragma unroll 8
  for (int d = 0; d < Dc; ++d)
    acc += xrow[d] * gw[d * Ec + e];

  __shared__ float lg[16][17];
  lg[tl][e] = acc;
  __syncthreads();

  float m = -3.4e38f;
  #pragma unroll
  for (int j = 0; j < Ec; ++j) m = fmaxf(m, lg[tl][j]);
  float sum = 0.0f;
  #pragma unroll
  for (int j = 0; j < Ec; ++j) sum += __expf(lg[tl][j] - m);

  aff[((size_t)b * Ec + e) * Sc + s] = __expf(acc - m) / sum;
}

// ---------------------------------------------------------------------------
// Kernel 3: per-(b,e) top-k (k=1024 of 8192) via in-LDS bitonic sort of
// packed monotonic keys.  Softmax values are > 0 so their IEEE bit pattern
// is order-preserving; ~index breaks ties toward the lower index (JAX top_k).
// Scatter-add combine makes the output order-insensitive, so an unordered
// top-k set is sufficient.
// ---------------------------------------------------------------------------
__global__ __launch_bounds__(1024) void moe_topk(const float* __restrict__ aff,
                                                 float* __restrict__ gating,
                                                 int* __restrict__ indexp) {
  __shared__ unsigned long long keys[Sc];   // 64 KB
  const int row = blockIdx.x;               // b*E + e
  const int tid = threadIdx.x;
  const float* a = aff + (size_t)row * Sc;

  for (int i = tid; i < Sc; i += 1024) {
    unsigned int vb = __builtin_bit_cast(unsigned int, a[i]);
    keys[i] = ((unsigned long long)vb << 32) | (unsigned int)(~i);
  }

  for (unsigned int k = 2; k <= Sc; k <<= 1) {
    for (unsigned int j = k >> 1; j > 0; j >>= 1) {
      __syncthreads();
      for (unsigned int i = tid; i < Sc; i += 1024) {
        unsigned int ixj = i ^ j;
        if (ixj > i) {
          unsigned long long va = keys[i], vb = keys[ixj];
          bool desc = ((i & k) == 0);
          if (desc ? (va < vb) : (va > vb)) { keys[i] = vb; keys[ixj] = va; }
        }
      }
    }
  }
  __syncthreads();

  if (tid < Kc) {
    unsigned long long kk = keys[tid];
    gating[(size_t)row * Kc + tid] =
        __builtin_bit_cast(float, (unsigned int)(kk >> 32));
    indexp[(size_t)row * Kc + tid] = (int)(~(unsigned int)kk);
  }
}

// ---------------------------------------------------------------------------
// Fused per-expert MLP: gather -> GEMM1 -> gelu -> GEMM2 -> gated scatter-add.
// 256 threads = 8 waves; wave w owns output columns [w*64, w*64+64).
// A fragments follow the CDNA5 16-bit 16x32 layout (two contiguous 8-element
// K runs per lane -> two 16B ds_load_b128); B fragments use lane->N,
// element->K in the same order, i.e. one bf16 from each of 16 weight rows per lane.
// ---------------------------------------------------------------------------
static __device__ __forceinline__ void gemm_64x512_bf16(
    const __bf16* __restrict__ lds,          // A tile [64,512] bf16
    const unsigned short* __restrict__ wB,   // B [512,512] bf16 row-major
    int w, int lane, v8f acc[4][4]) {
  const int base = (lane >= 16) ? 8 : 0;
  const int mloc = lane & 15;
  #pragma unroll 1
  for (int ks = 0; ks < 16; ++ks) {
    const int k0 = ks * 32;
    V16 a[4];
    #pragma unroll
    for (int mt = 0; mt < 4; ++mt) {
      const __bf16* ap = lds + (mt * 16 + mloc) * Dc + k0 + base;
      a[mt].h[0] = *(const v8bf*)ap;          // K = k0+base    .. +7
      a[mt].h[1] = *(const v8bf*)(ap + 16);   // K = k0+base+16 .. +7
    }
    #pragma unroll
    for (int nt = 0; nt < 4; ++nt) {
      const int ncol = w * 64 + nt * 16;
      const unsigned short* bp = wB + (size_t)(k0 + base) * 512 + ncol + mloc;
      V16 bb;                                  // lane -> N, element -> K (table order)
      #pragma unroll
      for (int s = 0; s < 8; ++s) {
        bb.v[s]     = __builtin_bit_cast(__bf16, bp[(size_t)s * 512]);
        bb.v[s + 8] = __builtin_bit_cast(__bf16, bp[(size_t)(s + 16) * 512]);
      }
      #pragma unroll
      for (int mt = 0; mt < 4; ++mt) {
        acc[mt][nt] = __builtin_amdgcn_wmma_f32_16x16x32_bf16(
            false, a[mt].v, false, bb.v, (short)0, acc[mt][nt], false, false);
      }
    }
  }
}

__global__ __launch_bounds__(256, 1) void moe_mlp(
    const float* __restrict__ x,
    const unsigned short* __restrict__ w1b,
    const float* __restrict__ b1,
    const unsigned short* __restrict__ w2b,
    const float* __restrict__ b2,
    const float* __restrict__ gating,
    const int* __restrict__ indexp,
    float* __restrict__ out) {
  __shared__ __bf16 tile[MT * Dc];    // 64 KB: A-tile, then reused for H

  const int wg   = blockIdx.x;
  const int mblk = wg & 15;           // which 64-slot block of K
  const int e    = (wg >> 4) & 15;
  const int b    = wg >> 8;

  const int tid  = threadIdx.x;
  const int lane = tid & 31;
  const int w    = tid >> 5;

  const size_t gi_base = ((size_t)b * Ec + e) * Kc + (size_t)mblk * MT;

  // ---- Stage A: gather 64 x-rows, convert to bf16 into LDS ----
  {
    const int r    = tid >> 2;            // 0..63
    const int col0 = (tid & 3) * 128;     // 4 threads per row
    const int tok  = indexp[gi_base + r];
    const float4* src =
        (const float4*)(x + ((size_t)b * Sc + tok) * Dc + col0);
    unsigned int* dst = (unsigned int*)&tile[r * Dc + col0];
    #pragma unroll
    for (int i = 0; i < 32; ++i) {
      float4 f = src[i];
      dst[2 * i]     = pack_bf16x2(f.x, f.y);
      dst[2 * i + 1] = pack_bf16x2(f.z, f.w);
    }
  }
  __syncthreads();

  v8f acc[4][4];
  const v8f vzero = {0.f, 0.f, 0.f, 0.f, 0.f, 0.f, 0.f, 0.f};
  #pragma unroll
  for (int mt = 0; mt < 4; ++mt)
    #pragma unroll
    for (int nt = 0; nt < 4; ++nt) acc[mt][nt] = vzero;

  // ---- GEMM1: h = x_in @ w1[e] ----
  gemm_64x512_bf16(tile, w1b + (size_t)e * Dc * DFFc, w, lane, acc);
  __syncthreads();                      // everyone done reading A-tile

  // ---- Epilogue 1: bias + gelu -> bf16 H tile (overwrite LDS) ----
  {
    const int nbase = w * 64 + (lane & 15);
    const int mh    = (lane >> 4) * 8;
    float bias[4];
    #pragma unroll
    for (int nt = 0; nt < 4; ++nt) bias[nt] = b1[e * DFFc + nbase + nt * 16];
    #pragma unroll
    for (int mt = 0; mt < 4; ++mt)
      #pragma unroll
      for (int nt = 0; nt < 4; ++nt)
        #pragma unroll
        for (int j = 0; j < 8; ++j) {
          float v = gelu_tanh(acc[mt][nt][j] + bias[nt]);
          tile[(mt * 16 + mh + j) * Dc + nbase + nt * 16] = (__bf16)v;
        }
  }
  __syncthreads();

  #pragma unroll
  for (int mt = 0; mt < 4; ++mt)
    #pragma unroll
    for (int nt = 0; nt < 4; ++nt) acc[mt][nt] = vzero;

  // ---- GEMM2: x_e = h @ w2[e] ----
  gemm_64x512_bf16(tile, w2b + (size_t)e * DFFc * Dc, w, lane, acc);

  // ---- Epilogue 2: bias, gate-scale, atomic scatter-add into out ----
  {
    const int nbase = w * 64 + (lane & 15);
    const int mh    = (lane >> 4) * 8;
    float bias[4];
    #pragma unroll
    for (int nt = 0; nt < 4; ++nt) bias[nt] = b2[e * Dc + nbase + nt * 16];
    #pragma unroll
    for (int mt = 0; mt < 4; ++mt) {
      #pragma unroll
      for (int j = 0; j < 8; ++j) {
        const int m   = mt * 16 + mh + j;
        const float g = gating[gi_base + m];
        const int tok = indexp[gi_base + m];
        float* orow = out + ((size_t)b * Sc + tok) * Dc;
        #pragma unroll
        for (int nt = 0; nt < 4; ++nt)
          atomicAdd(orow + nbase + nt * 16, (acc[mt][nt][j] + bias[nt]) * g);
      }
    }
  }
}

// ---------------------------------------------------------------------------
extern "C" void kernel_launch(void* const* d_in, const int* in_sizes, int n_in,
                              void* d_out, int out_size, void* d_ws,
                              size_t ws_size, hipStream_t stream) {
  const float* x  = (const float*)d_in[0];
  const float* gw = (const float*)d_in[1];
  const float* w1 = (const float*)d_in[2];
  const float* b1 = (const float*)d_in[3];
  const float* w2 = (const float*)d_in[4];
  const float* b2 = (const float*)d_in[5];
  float* out = (float*)d_out;

  // Workspace layout (all 16B-aligned):
  //   aff   [B,E,S]  f32   4 MB
  //   gate  [B,E,K]  f32   0.5 MB
  //   index [B,E,K]  i32   0.5 MB
  //   w1b   [E,D,DFF] bf16 8 MB
  //   w2b   [E,DFF,D] bf16 8 MB
  char* p = (char*)d_ws;
  float*          aff    = (float*)p;           p += (size_t)Bc * Ec * Sc * 4;
  float*          gating = (float*)p;           p += (size_t)Bc * Ec * Kc * 4;
  int*            indexp = (int*)p;             p += (size_t)Bc * Ec * Kc * 4;
  unsigned short* w1b    = (unsigned short*)p;  p += (size_t)Ec * Dc * DFFc * 2;
  unsigned short* w2b    = (unsigned short*)p;

  hipMemsetAsync(d_out, 0, (size_t)out_size * sizeof(float), stream);

  {
    const int n = Ec * Dc * DFFc;
    moe_cvt_weights<<<(n + 255) / 256, 256, 0, stream>>>(w1, w2, w1b, w2b);
  }
  moe_router<<<Bc * (Sc / 16), 256, 0, stream>>>(x, gw, aff);
  moe_topk<<<Bc * Ec, 1024, 0, stream>>>(aff, gating, indexp);
  moe_mlp<<<Bc * Ec * (Kc / MT), 256, 0, stream>>>(x, w1b, b1, w2b, b2,
                                                   gating, indexp, out);
}
